// MixtralMoE_55147380081301
// MI455X (gfx1250) — matrix operand order repaired and checked by reading, not compiled, not run on hardware
//
#include <hip/hip_runtime.h>
#include <hip/hip_bf16.h>
#include <cstddef>

// ---------------- problem constants ----------------
#define E_    8
#define TOPK_ 2
#define H_    2048
#define F_    4096
#define T_    8192

#define MTILE 32                    // tokens per GEMM block (two 16-row A fragments)
#define MAX_TILES 520               // T*2/32 + E
#define MAX_SLOTS (MAX_TILES * 32)  // 16640 padded token-expert slots

typedef __bf16 bf16_t;
typedef __bf16 v16bf __attribute__((ext_vector_type(16)));
typedef __bf16 v8bf  __attribute__((ext_vector_type(8)));
typedef float  v8f   __attribute__((ext_vector_type(8)));
typedef float  v4f   __attribute__((ext_vector_type(4)));

// ---------------- workspace layout (bytes) ----------------
static constexpr size_t OFF_COUNTS = 0;                                   // E ints
static constexpr size_t OFF_FILL   = 64;                                  // E ints
static constexpr size_t OFF_PADOFF = 128;                                 // E+1 ints
static constexpr size_t OFF_NTILES = 192;                                 // 1 int
static constexpr size_t OFF_TEXP   = 256;                                 // MAX_TILES ints
static constexpr size_t OFF_TBASE  = OFF_TEXP + sizeof(int) * MAX_TILES;
static constexpr size_t OFF_STOK   = ((OFF_TBASE + sizeof(int) * MAX_TILES) + 255) & ~size_t(255);
static constexpr size_t OFF_SW     = OFF_STOK + sizeof(int) * MAX_SLOTS;
static constexpr size_t OFF_TKID   = OFF_SW + sizeof(float) * MAX_SLOTS;
static constexpr size_t OFF_TKW    = OFF_TKID + sizeof(int) * T_ * 2;
static constexpr size_t OFF_HWS    = ((OFF_TKW + sizeof(float) * T_ * 2) + 511) & ~size_t(511);
static constexpr size_t HWS_BYTES  = (size_t)MAX_SLOTS * F_ * 2;          // h, A-fragment order
static constexpr size_t WMAT_BYTES = (size_t)E_ * F_ * H_ * 2;            // one packed weight set
static constexpr size_t OFF_W1S    = ((OFF_HWS + HWS_BYTES) + 511) & ~size_t(511);
static constexpr size_t OFF_W3S    = OFF_W1S + WMAT_BYTES;
static constexpr size_t OFF_W2S    = OFF_W3S + WMAT_BYTES;

// ---------------- helpers ----------------
__device__ __forceinline__ v16bf combine8(v8bf lo, v8bf hi) {
    v16bf r;
#pragma unroll
    for (int i = 0; i < 8; ++i) { r[i] = lo[i]; r[i + 8] = hi[i]; }
    return r;
}

__device__ __forceinline__ v16bf load16_f32_to_bf16(const float* __restrict__ p) {
    const v4f* q = (const v4f*)p;
    v4f a = q[0], b = q[1], c = q[2], d = q[3];
    v16bf r;
#pragma unroll
    for (int i = 0; i < 4; ++i) {
        r[i]      = (bf16_t)a[i];
        r[i + 4]  = (bf16_t)b[i];
        r[i + 8]  = (bf16_t)c[i];
        r[i + 12] = (bf16_t)d[i];
    }
    return r;
}

// silu(g) = g * sigmoid(g), sigmoid via single v_rcp_f32
__device__ __forceinline__ float silu_fast(float g) {
    return g * __builtin_amdgcn_rcpf(1.f + __expf(-g));
}

#define WMMA_BF16(A, B, C) \
    __builtin_amdgcn_wmma_f32_16x16x32_bf16(false, (A), false, (B), (short)0, (C), false, false)

// ---------------- K0: router ----------------
__global__ __launch_bounds__(256) void moe_router(const float* __restrict__ x,
                                                  const float* __restrict__ gw,
                                                  int* __restrict__ counts,
                                                  int* __restrict__ tkid,
                                                  float* __restrict__ tkw) {
    const int wave = threadIdx.x >> 5;
    const int lane = threadIdx.x & 31;
    const int t = blockIdx.x * 8 + wave;
    if (t >= T_) return;

    const float* xr = x + (size_t)t * H_;
    float acc[E_];
#pragma unroll
    for (int e = 0; e < E_; ++e) acc[e] = 0.f;

    for (int i = lane; i < H_; i += 32) {
        float xv = xr[i];
#pragma unroll
        for (int e = 0; e < E_; ++e) acc[e] += xv * gw[e * H_ + i];
    }
#pragma unroll
    for (int e = 0; e < E_; ++e) {
#pragma unroll
        for (int off = 16; off > 0; off >>= 1) acc[e] += __shfl_xor(acc[e], off, 32);
    }

    if (lane == 0) {
        float mx = acc[0];
#pragma unroll
        for (int e = 1; e < E_; ++e) mx = fmaxf(mx, acc[e]);
        float p[E_];
#pragma unroll
        for (int e = 0; e < E_; ++e) p[e] = __expf(acc[e] - mx);
        int i0 = 0;
#pragma unroll
        for (int e = 1; e < E_; ++e) if (p[e] > p[i0]) i0 = e;
        int i1 = (i0 == 0) ? 1 : 0;
#pragma unroll
        for (int e = 0; e < E_; ++e) if (e != i0 && p[e] > p[i1]) i1 = e;
        float inv = 1.f / (p[i0] + p[i1]);
        tkid[2 * t + 0] = i0; tkw[2 * t + 0] = p[i0] * inv;
        tkid[2 * t + 1] = i1; tkw[2 * t + 1] = p[i1] * inv;
        atomicAdd(&counts[i0], 1);
        atomicAdd(&counts[i1], 1);
    }
}

// ---------------- K1: scan counts, build 32-row tile tables ----------------
__global__ __launch_bounds__(256) void moe_setup(const int* __restrict__ counts,
                                                 int* __restrict__ fill,
                                                 int* __restrict__ pad_off,
                                                 int* __restrict__ ntiles,
                                                 int* __restrict__ texp,
                                                 int* __restrict__ tbase,
                                                 int* __restrict__ stok) {
    __shared__ int s_nt;
    if (threadIdx.x == 0) {
        int off = 0, nt = 0;
        for (int e = 0; e < E_; ++e) {
            pad_off[e] = off;
            int c = counts[e];
            int nti = (c + MTILE - 1) / MTILE;
            for (int j = 0; j < nti; ++j) { texp[nt] = e; tbase[nt] = off + j * MTILE; ++nt; }
            off += nti * MTILE;
            fill[e] = 0;
        }
        pad_off[E_] = off;
        *ntiles = nt;
        s_nt = nt;
    }
    __syncthreads();
    const int nt = s_nt;
    for (int i = threadIdx.x; i < MAX_TILES; i += blockDim.x)
        if (i >= nt) texp[i] = -1;
    for (int i = threadIdx.x; i < MAX_SLOTS; i += blockDim.x)
        stok[i] = -1;
}

// ---------------- K2: scatter ----------------
__global__ __launch_bounds__(256) void moe_scatter(const int* __restrict__ tkid,
                                                   const float* __restrict__ tkw,
                                                   int* __restrict__ fill,
                                                   const int* __restrict__ pad_off,
                                                   int* __restrict__ stok,
                                                   float* __restrict__ sw) {
    int t = blockIdx.x * blockDim.x + threadIdx.x;
    if (t >= T_) return;
#pragma unroll
    for (int k = 0; k < TOPK_; ++k) {
        int e = tkid[2 * t + k];
        int pos = atomicAdd(&fill[e], 1);
        int slot = pad_off[e] + pos;
        stok[slot] = t;
        sw[slot] = tkw[2 * t + k];
    }
}

// ---------------- KP: pack f32 weights -> bf16 B-fragment order, k = 16*half + s ----------------
__global__ __launch_bounds__(256) void pack_wb(const float* __restrict__ src,
                                               bf16_t* __restrict__ dst,
                                               int R, int C) {
    const int lane = threadIdx.x & 31;
    const int frag = blockIdx.x * 8 + (threadIdx.x >> 5);
    const int ktiles = C >> 5;
    const int ftile = frag / ktiles;
    const int ktile = frag % ktiles;
    const int n = lane & 15, khalf = lane >> 4;
    const float* s = src + (size_t)(ftile * 16 + n) * C + ktile * 32 + khalf * 16;
    *(v16bf*)(dst + (size_t)frag * 512 + lane * 16) = load16_f32_to_bf16(s);
}

// ---------------- K3: h = silu(x@w1^T) * (x@w3^T), 32-token x 32-F tile per wave step ----------------
// Per k-step: 2 LDS A-frags + 4 global B-frags -> 8 WMMAs (B reused across 2 M-tiles).
__global__ __launch_bounds__(256, 1) void moe_h_gemm(const float* __restrict__ x,
                                                     const bf16_t* __restrict__ w1s,
                                                     const bf16_t* __restrict__ w3s,
                                                     const int* __restrict__ texp,
                                                     const int* __restrict__ tbase,
                                                     const int* __restrict__ stok,
                                                     bf16_t* __restrict__ hws) {
    __shared__ bf16_t xs[MTILE][H_ + 8];     // 131.5KB; +8 pad -> conflict-free A loads
    __shared__ bf16_t stage[8][2][16][24];   // per-wave staging for one 16x32 h tile
    __shared__ int    s_tok[MTILE];

    const int ex = texp[blockIdx.x];
    if (ex < 0) return;
    const int base = tbase[blockIdx.x];

    if (threadIdx.x < MTILE) s_tok[threadIdx.x] = stok[base + threadIdx.x];
    __syncthreads();

    // Gather 32 token rows of x (f32) -> LDS bf16. 8 threads per row, 256 cols each.
    {
        const int r  = threadIdx.x >> 3;
        const int c0 = (threadIdx.x & 7) * (H_ / 8);
        const int tok = s_tok[r];
        if (tok >= 0) {
            const v4f* src = (const v4f*)(x + (size_t)tok * H_ + c0);
            for (int i = 0; i < (H_ / 8) / 4; ++i) {
                v4f v = src[i];
                int c = c0 + i * 4;
#pragma unroll
                for (int j = 0; j < 4; ++j) xs[r][c + j] = (bf16_t)v[j];
            }
        } else {
            for (int c = c0; c < c0 + H_ / 8; ++c) xs[r][c] = (bf16_t)0.f;
        }
    }
    __syncthreads();

    const int lane = threadIdx.x & 31;
    const int wave = threadIdx.x >> 5;
    const int n    = lane & 15;
    const int hi   = lane >> 4;
    const int akb  = hi * 8;

    const bf16_t* arow0 = &xs[n][0];        // A rows 0..15
    const bf16_t* arow1 = &xs[n + 16][0];   // A rows 16..31
    const size_t laneoff = (size_t)lane * 16;
    const size_t KT = H_ / 32;

    // SwiGLU + transpose-through-LDS + fragment-order store of one 16x32 h tile
    auto emit_tile = [&](int mi, const v8f& g0, const v8f& u0, const v8f& g1, const v8f& u1, int f0) {
#pragma unroll
        for (int r = 0; r < 8; ++r) {
            stage[wave][0][r + akb][n] = (bf16_t)(silu_fast(g0[r]) * u0[r]);
            stage[wave][1][r + akb][n] = (bf16_t)(silu_fast(g1[r]) * u1[r]);
        }
        const int m = lane & 15;
        v8bf pa = *(const v8bf*)&stage[wave][hi][m][0];
        v8bf pb = *(const v8bf*)&stage[wave][hi][m][8];
        *(v16bf*)(hws + ((size_t)((base >> 4) + mi) * (F_ / 32) + (f0 >> 5)) * 512 + laneoff) =
            combine8(pa, pb);
    };

    for (int ntp = wave * 2; ntp < F_ / 16; ntp += 16) {
        const int f0 = ntp * 16;
        const bf16_t* b1a = w1s + ((size_t)(ex * (F_ / 16) + ntp) * KT) * 512 + laneoff;
        const bf16_t* b3a = w3s + ((size_t)(ex * (F_ / 16) + ntp) * KT) * 512 + laneoff;
        const bf16_t* b1b = b1a + KT * 512;
        const bf16_t* b3b = b3a + KT * 512;

        v8f c1a0, c3a0, c1b0, c3b0, c1a1, c3a1, c1b1, c3b1;
#pragma unroll
        for (int r = 0; r < 8; ++r) {
            c1a0[r] = 0.f; c3a0[r] = 0.f; c1b0[r] = 0.f; c3b0[r] = 0.f;
            c1a1[r] = 0.f; c3a1[r] = 0.f; c1b1[r] = 0.f; c3b1[r] = 0.f;
        }

        for (int kt = 0; kt < (int)KT; ++kt) {
            const size_t fo = (size_t)kt * 512;
            __builtin_prefetch(b1a + fo + 4096, 0, 1);
            __builtin_prefetch(b3a + fo + 4096, 0, 1);
            const int k0 = kt * 32;
            v16bf A0 = combine8(*(const v8bf*)(arow0 + k0 + 2 * akb),
                                *(const v8bf*)(arow0 + k0 + 2 * akb + 8));
            v16bf A1 = combine8(*(const v8bf*)(arow1 + k0 + 2 * akb),
                                *(const v8bf*)(arow1 + k0 + 2 * akb + 8));
            v16bf B1a = *(const v16bf*)(b1a + fo);
            v16bf B3a = *(const v16bf*)(b3a + fo);
            v16bf B1b = *(const v16bf*)(b1b + fo);
            v16bf B3b = *(const v16bf*)(b3b + fo);
            c1a0 = WMMA_BF16(A0, B1a, c1a0);
            c1a1 = WMMA_BF16(A1, B1a, c1a1);
            c3a0 = WMMA_BF16(A0, B3a, c3a0);
            c3a1 = WMMA_BF16(A1, B3a, c3a1);
            c1b0 = WMMA_BF16(A0, B1b, c1b0);
            c1b1 = WMMA_BF16(A1, B1b, c1b1);
            c3b0 = WMMA_BF16(A0, B3b, c3b0);
            c3b1 = WMMA_BF16(A1, B3b, c3b1);
        }

        emit_tile(0, c1a0, c3a0, c1b0, c3b0, f0);
        emit_tile(1, c1a1, c3a1, c1b1, c3b1, f0);
    }
}

// ---------------- K4: out[t] += w_e * (h @ w2^T), 32-token M-tile ----------------
// Per k-step: 2 A-frags + 2 B-frags -> 4 WMMAs, all coalesced v16bf loads.
__global__ __launch_bounds__(256, 1) void moe_out_gemm(const bf16_t* __restrict__ hws,
                                                       const bf16_t* __restrict__ w2s,
                                                       const int* __restrict__ texp,
                                                       const int* __restrict__ tbase,
                                                       const int* __restrict__ stok,
                                                       const float* __restrict__ sw,
                                                       float* __restrict__ out) {
    __shared__ int   s_tok[MTILE];
    __shared__ float s_w[MTILE];

    const int ex = texp[blockIdx.x];
    if (ex < 0) return;
    const int base = tbase[blockIdx.x];

    if (threadIdx.x < MTILE) {
        s_tok[threadIdx.x] = stok[base + threadIdx.x];
        s_w[threadIdx.x]   = sw[base + threadIdx.x];
    }
    __syncthreads();

    const int lane = threadIdx.x & 31;
    const int wave = threadIdx.x >> 5;
    const int n    = lane & 15;
    const int hi   = lane >> 4;
    const int akb  = hi * 8;

    int   tok0[8], tok1[8];
    float wr0[8], wr1[8];
#pragma unroll
    for (int r = 0; r < 8; ++r) {
        tok0[r] = s_tok[r + akb];      wr0[r] = s_w[r + akb];
        tok1[r] = s_tok[16 + r + akb]; wr1[r] = s_w[16 + r + akb];
    }

    const size_t laneoff = (size_t)lane * 16;
    const size_t KT = F_ / 32;
    const bf16_t* af0 = hws + (size_t)(base >> 4) * KT * 512 + laneoff;
    const bf16_t* af1 = af0 + KT * 512;

    for (int ntp = wave * 2; ntp < H_ / 16; ntp += 16) {
        const int c0 = ntp * 16;
        const int c1 = c0 + 16;
        const bf16_t* b0 = w2s + ((size_t)(ex * (H_ / 16) + ntp) * KT) * 512 + laneoff;
        const bf16_t* b1 = b0 + KT * 512;

        v8f a00, a01, a10, a11;   // [mtile][ntile]
#pragma unroll
        for (int r = 0; r < 8; ++r) { a00[r] = 0.f; a01[r] = 0.f; a10[r] = 0.f; a11[r] = 0.f; }

        for (int kt = 0; kt < (int)KT; ++kt) {
            const size_t fo = (size_t)kt * 512;
            __builtin_prefetch(b0 + fo + 4096, 0, 1);
            __builtin_prefetch(af0 + fo + 4096, 0, 1);
            v16bf A0 = *(const v16bf*)(af0 + fo);
            v16bf A1 = *(const v16bf*)(af1 + fo);
            v16bf B0 = *(const v16bf*)(b0 + fo);
            v16bf B1 = *(const v16bf*)(b1 + fo);
            a00 = WMMA_BF16(A0, B0, a00);
            a10 = WMMA_BF16(A1, B0, a10);
            a01 = WMMA_BF16(A0, B1, a01);
            a11 = WMMA_BF16(A1, B1, a11);
        }
#pragma unroll
        for (int r = 0; r < 8; ++r) {
            if (tok0[r] >= 0) {
                atomicAdd(&out[(size_t)tok0[r] * H_ + c0 + n], a00[r] * wr0[r]);
                atomicAdd(&out[(size_t)tok0[r] * H_ + c1 + n], a01[r] * wr0[r]);
            }
            if (tok1[r] >= 0) {
                atomicAdd(&out[(size_t)tok1[r] * H_ + c0 + n], a10[r] * wr1[r]);
                atomicAdd(&out[(size_t)tok1[r] * H_ + c1 + n], a11[r] * wr1[r]);
            }
        }
    }
}

// ---------------- host launcher ----------------
extern "C" void kernel_launch(void* const* d_in, const int* in_sizes, int n_in,
                              void* d_out, int out_size, void* d_ws, size_t ws_size,
                              hipStream_t stream) {
    const float* x  = (const float*)d_in[0];
    const float* gw = (const float*)d_in[1];
    const float* w1 = (const float*)d_in[2];
    const float* w3 = (const float*)d_in[3];
    const float* w2 = (const float*)d_in[4];
    float* out = (float*)d_out;

    char* ws = (char*)d_ws;
    int*    counts  = (int*)  (ws + OFF_COUNTS);
    int*    fill    = (int*)  (ws + OFF_FILL);
    int*    pad_off = (int*)  (ws + OFF_PADOFF);
    int*    ntiles  = (int*)  (ws + OFF_NTILES);
    int*    texp    = (int*)  (ws + OFF_TEXP);
    int*    tbase   = (int*)  (ws + OFF_TBASE);
    int*    stok    = (int*)  (ws + OFF_STOK);
    float*  slotw   = (float*)(ws + OFF_SW);
    int*    tkid    = (int*)  (ws + OFF_TKID);
    float*  tkw     = (float*)(ws + OFF_TKW);
    bf16_t* hws     = (bf16_t*)(ws + OFF_HWS);
    bf16_t* w1s     = (bf16_t*)(ws + OFF_W1S);
    bf16_t* w3s     = (bf16_t*)(ws + OFF_W3S);
    bf16_t* w2s     = (bf16_t*)(ws + OFF_W2S);

    hipMemsetAsync(counts, 0, 64, stream);
    hipMemsetAsync(out, 0, sizeof(float) * T_ * H_, stream);

    const int packBlocks = (E_ * (F_ / 16) * (H_ / 32)) / 8;     // 16384
    pack_wb<<<packBlocks, 256, 0, stream>>>(w1, w1s, F_, H_);
    pack_wb<<<packBlocks, 256, 0, stream>>>(w3, w3s, F_, H_);
    pack_wb<<<packBlocks, 256, 0, stream>>>(w2, w2s, H_, F_);

    moe_router  <<<T_ / 8,   256, 0, stream>>>(x, gw, counts, tkid, tkw);
    moe_setup   <<<1,        256, 0, stream>>>(counts, fill, pad_off, ntiles, texp, tbase, stok);
    moe_scatter <<<T_ / 256, 256, 0, stream>>>(tkid, tkw, fill, pad_off, stok, slotw);
    moe_h_gemm  <<<MAX_TILES, 256, 0, stream>>>(x, w1s, w3s, texp, tbase, stok, hws);
    moe_out_gemm<<<MAX_TILES, 256, 0, stream>>>(hws, w2s, texp, tbase, stok, slotw, out);
}
